// Model_25074019074663
// MI455X (gfx1250) — matrix operand order repaired and checked by reading, not compiled, not run on hardware
//
#include <hip/hip_runtime.h>

typedef __attribute__((ext_vector_type(16))) _Float16 v16h;
typedef __attribute__((ext_vector_type(8)))  _Float16 v8h;
typedef __attribute__((ext_vector_type(8)))  float    v8f;
typedef int v4i __attribute__((ext_vector_type(4)));

#define TV 25        // joints
#define TT 8         // t-rows per tile (== number of waves)
#define P  200       // positions per tile = TT*TV
#define PR 208       // padded row count for transposed tiles (chunk tail)
#define CSX 72       // xh_t stride in halfs (row = 144B, 16B aligned)
#define CSD 24       // xd_t stride in halfs (row = 48B, 16B aligned)
#define QS  264      // q_s row stride in halfs (528B: 16B aligned, bank-spread)
#define NCHUNK 13    // ceil(P/16)

#define AS_GLOBAL __attribute__((address_space(1)))
#define AS_LDS    __attribute__((address_space(3)))

#if defined(__has_builtin)
#if __has_builtin(__builtin_amdgcn_global_load_async_to_lds_b128) && \
    __has_builtin(__builtin_amdgcn_s_wait_asynccnt)
#define USE_ASYNC_LDS 1
#endif
#endif

#if defined(USE_ASYNC_LDS)
// async DMA of 16 bytes: global -> LDS (tracked by ASYNCcnt)
static __device__ __forceinline__ void async_copy16(const float* g, float* l) {
  __builtin_amdgcn_global_load_async_to_lds_b128(
      (AS_GLOBAL v4i*)(AS_GLOBAL void*)(void*)const_cast<float*>(g),
      (AS_LDS    v4i*)(AS_LDS    void*)(void*)l, 0, 0);
}
#endif

// ---------------------------------------------------------------------------
// Pass 1: fully WMMA-based fused block.
//   stage 1: xd = relu(bn(Wd[16x64] * X[64,P]))               (2 WMMA / chunk)
//   stage 2: q  = (s∘Wsub)[16x16] * xd                        (1 WMMA / chunk)
//   stage 3: Y_t = Q_t[16x32pad] * A^T[32pad x 16] per t      (2 WMMA / wave)
// stage-3 accumulates over layers in WMMA C registers; bias/bn constants are
// folded into per-channel S0/C0. Channels 0..47 written from registers.
// ---------------------------------------------------------------------------
__global__ __launch_bounds__(256) void fused_main(
    const float* __restrict__ x, const float* __restrict__ Aadj,
    const float* __restrict__ w_down, const float* __restrict__ b_down,
    const float* __restrict__ g_down, const float* __restrict__ bt_down,
    const float* __restrict__ w_sub,  const float* __restrict__ b_sub,
    const float* __restrict__ g_sub,  const float* __restrict__ bt_sub,
    const float* __restrict__ g_bn,   const float* __restrict__ b_bn,
    float* __restrict__ xm, float* __restrict__ out)
{
  extern __shared__ float smem[];
  float*    xs   = smem;                    // 64*200 f32 (x tile, residual)
  float*    A_s  = xs + 64 * P;             // 625    f32 (adjacency)
  float*    xm_s = A_s + 625;               // 400+3  f32 (T-partials + pad)
  float*    S0_s = xm_s + 403;              // 48 f32 (final bn scale)
  float*    C0_s = S0_s + 48;               // 48 f32 (folded bias constants)
  _Float16* xh_t = (_Float16*)(C0_s + 48);  // PR*CSX f16, pos-major x
  _Float16* xd_t = xh_t + PR * CSX;         // PR*CSD f16, pos-major xd
  _Float16* q_s  = xd_t + PR * CSD;         // 16*QS  f16, channel-mixed (padded)

  const int tid  = threadIdx.x;
  const int lane = tid & 31;
  const int wave = tid >> 5;
  const int hi   = lane >> 4;     // lane half (WMMA fragment layout)
  const int mrow = lane & 15;
  const int n    = blockIdx.y;
  const int t0   = blockIdx.x * TT;
  const size_t xbase = (size_t)n * 64 * 6400 + (size_t)t0 * TV;
  const float RS = rsqrtf(1.0f + 1e-5f);

  // ---- stage x tile into LDS (async DMA path when available) --------------
#if defined(USE_ASYNC_LDS)
  for (int idx = tid * 4; idx < 64 * P; idx += 256 * 4) {
    int c = idx / P, p = idx % P;   // rows of 200 floats, p multiple of 4
    async_copy16(x + xbase + (size_t)c * 6400 + p, xs + c * P + p);
  }
#else
  for (int idx = tid; idx < 64 * P; idx += 256) {
    int c = idx / P, p = idx % P;
    xs[c * P + p] = x[xbase + (size_t)c * 6400 + p];
  }
#endif
  // zero padded q staging once; stores never touch u>=25 so pads stay zero
  for (int idx = tid; idx < 16 * QS; idx += 256) q_s[idx] = (_Float16)0.f;
  // folded per-channel constants for channels 0..47:
  //   S0 = g_bn/sqrt(1+eps);  C0 = (sum_li b_sub*s_sub + bt_sub)*S0 + b_bn
  if (tid < 48) {
    int j = tid / 16, o = tid % 16;
    float bsum = 0.f;
    for (int li = 0; li < 3; ++li) {
      int pi = (li * 3 + j) * 16 + o;
      bsum += b_sub[pi] * (g_sub[pi] * RS) + bt_sub[pi];
    }
    float sbn = g_bn[tid] * RS;
    S0_s[tid] = sbn;
    C0_s[tid] = bsum * sbn + b_bn[tid];
  }
#if defined(USE_ASYNC_LDS)
  __builtin_amdgcn_s_wait_asynccnt(0);
#endif
  __syncthreads();

  // one-time f32 -> f16 transpose: xh_t[p][c] feeds WMMA B-fragments
  for (int idx = tid; idx < 64 * P; idx += 256) {
    int c = idx / P, p = idx % P;
    xh_t[p * CSX + c] = (_Float16)xs[c * P + p];
  }
  __syncthreads();

  // per-wave register accumulators: [subset j][v-chunk], chained as WMMA C
  v8f yacc[3][2] = {};
  const int tw = wave;              // this wave owns t-row tw of the tile

  for (int li = 0; li < 3; ++li) {
    for (int idx = tid; idx < 400; idx += 256) xm_s[idx] = 0.f;
    __syncthreads();

    // ---- stage 1: xd[16,P] = relu(bn(Wd[16x64] * X[64,P])) ----------------
    const float* wd = w_down + li * 16 * 64;
    v16h a0, a1;
#pragma unroll
    for (int h = 0; h < 16; ++h) {
      int K = (hi << 4) + h;   // contiguous k = 16*hi + h, same order as B
      a0[h] = (_Float16)wd[mrow * 64 + K];
      a1[h] = (_Float16)wd[mrow * 64 + 32 + K];
    }
    float sd[8], bd[8], btd[8];       // hoisted bn params, rows m = r + hi*8
#pragma unroll
    for (int r = 0; r < 8; ++r) {
      int pi = li * 16 + r + (hi << 3);
      sd[r] = g_down[pi] * RS; bd[r] = b_down[pi]; btd[r] = bt_down[pi];
    }
    for (int ch = wave; ch < NCHUNK; ch += 8) {
      int p = ch * 16 + mrow;          // this lane's B/D column position
      const _Float16* rowp = xh_t + p * CSX + (hi << 4);
      v8h t0v = *(const v8h*)(rowp);
      v8h t1v = *(const v8h*)(rowp + 8);
      v8h t2v = *(const v8h*)(rowp + 32);
      v8h t3v = *(const v8h*)(rowp + 40);
      v16h b0, b1;
#pragma unroll
      for (int h = 0; h < 8; ++h) {
        b0[h] = t0v[h]; b0[h + 8] = t1v[h];
        b1[h] = t2v[h]; b1[h + 8] = t3v[h];
      }
      v8f d = {};
      d = __builtin_amdgcn_wmma_f32_16x16x32_f16(false, a0, false, b0, (short)0, d, false, false);
      d = __builtin_amdgcn_wmma_f32_16x16x32_f16(false, a1, false, b1, (short)0, d, false, false);
      if (p < P) {
        int v = p % TV;
#pragma unroll
        for (int r = 0; r < 8; ++r) {
          float val = fmaxf((d[r] + bd[r]) * sd[r] + btd[r], 0.f);
          xd_t[p * CSD + r + (hi << 3)] = (_Float16)val;
          atomicAdd(&xm_s[(r + (hi << 3)) * TV + v], val);  // edge T-partial
        }
      }
    }
    __syncthreads();

    // flush T-partials (edge branch needs mean over all T)
    for (int idx = tid; idx < 400; idx += 256)
      atomicAdd(&xm[((size_t)li * 64 + n) * 400 + idx], xm_s[idx]);

    // ---- stages 2+3 per subset --------------------------------------------
#pragma unroll
    for (int j = 0; j < 3; ++j) {
      for (int idx = tid; idx < 625; idx += 256)
        A_s[idx] = Aadj[(li * 3 + j) * 625 + idx];
      const float* ws  = w_sub + (li * 3 + j) * 256;
      const float  ssc = g_sub[(li * 3 + j) * 16 + mrow] * RS; // bn fold, row mrow
      v16h aw;
#pragma unroll
      for (int h = 0; h < 16; ++h) {   // K padded 16 -> 32 with zeros
        int K = (hi << 4) + h;
        aw[h] = (K < 16) ? (_Float16)(ws[mrow * 16 + K] * ssc) : (_Float16)0.f;
      }
      __syncthreads();                 // q_s free to overwrite; A_s ready
      for (int ch = wave; ch < NCHUNK; ch += 8) {
        int p = ch * 16 + mrow;
        v16h bq = {};                  // hi lanes carry K>=16 -> zeros
        if (hi == 0) {
          const _Float16* rp = xd_t + p * CSD;
          v8h u0 = *(const v8h*)(rp);
          v8h u1 = *(const v8h*)(rp + 8);
#pragma unroll
          for (int h = 0; h < 8; ++h) { bq[h] = u0[h]; bq[h + 8] = u1[h]; }
        }
        v8f d = {};
        d = __builtin_amdgcn_wmma_f32_16x16x32_f16(false, aw, false, bq, (short)0, d, false, false);
        if (p < P) {
          int tq = p / TV, uq = p % TV;
#pragma unroll
          for (int r = 0; r < 8; ++r)
            q_s[(r + (hi << 3)) * QS + tq * 32 + uq] = (_Float16)d[r];
        }
      }
      __syncthreads();
      // stage 3: Y_t += Q_t x A^T as WMMA; this wave handles t = tw
      v16h bA0, bA1;
#pragma unroll
      for (int h = 0; h < 16; ++h) {   // B-frag: K=u = (h/8)*16+hi*8+h%8, col v
        int u = ((h >> 3) << 4) + (hi << 3) + (h & 7);
        bA0[h] = (u < TV) ? (_Float16)A_s[mrow * TV + u] : (_Float16)0.f;
        bA1[h] = (u < TV && mrow < 9) ? (_Float16)A_s[(16 + mrow) * TV + u]
                                      : (_Float16)0.f;
      }
      const _Float16* qp = q_s + mrow * QS + tw * 32;
      v8h qa = *(const v8h*)(qp + (hi << 3));        // u = hi*8 + 0..7
      v8h qb = *(const v8h*)(qp + 16 + (hi << 3));   // u = 16 + hi*8 + 0..7
      v16h aq;
#pragma unroll
      for (int h = 0; h < 8; ++h) { aq[h] = qa[h]; aq[h + 8] = qb[h]; }
      yacc[j][0] = __builtin_amdgcn_wmma_f32_16x16x32_f16(false, aq, false, bA0,
                                                          (short)0, yacc[j][0], false, false);
      yacc[j][1] = __builtin_amdgcn_wmma_f32_16x16x32_f16(false, aq, false, bA1,
                                                          (short)0, yacc[j][1], false, false);
      __syncthreads();
    }
  }

  // ---- final: out[ch 0..47] = relu(yacc*S0 + C0 + x) straight from regs ---
#pragma unroll
  for (int j = 0; j < 3; ++j) {
#pragma unroll
    for (int cnk = 0; cnk < 2; ++cnk) {
      int v = mrow + cnk * 16;
      if (v < TV) {
#pragma unroll
        for (int r = 0; r < 8; ++r) {
          int oc = j * 16 + r + (hi << 3);
          float val = yacc[j][cnk][r] * S0_s[oc] + C0_s[oc]
                    + xs[oc * P + tw * TV + v];
          out[xbase + (size_t)oc * 6400 + tw * TV + v] = fmaxf(val, 0.f);
        }
      }
    }
  }
}

// ---------------------------------------------------------------------------
// Pass 2: edge conv on the T-mean (tiny). One block per n; sums over layers.
// ---------------------------------------------------------------------------
__global__ __launch_bounds__(128) void edge_conv_kernel(
    const float* __restrict__ xm, const float* __restrict__ w_edge,
    const float* __restrict__ g_edge, const float* __restrict__ bt_edge,
    float* __restrict__ hsum)
{
  __shared__ float xt[25 * 16];
  __shared__ float xx[25];
  __shared__ int   sidx[25 * 5];
  __shared__ float hk[16 * 25 * 5];
  __shared__ float hacc[16 * 25];
  const int n = blockIdx.x, tid = threadIdx.x;
  const float RS = rsqrtf(1.0f + 1e-5f);

  for (int i = tid; i < 400; i += 128) hacc[i] = 0.f;

  for (int li = 0; li < 3; ++li) {
    __syncthreads();
    for (int i = tid; i < 400; i += 128) {
      int v = i / 16, c = i % 16;
      xt[v * 16 + c] = xm[((size_t)li * 64 + n) * 400 + c * 25 + v] * (1.0f / 256.0f);
    }
    __syncthreads();
    if (tid < 25) {
      float s = 0.f;
      for (int c = 0; c < 16; ++c) { float a = xt[tid * 16 + c]; s += a * a; }
      xx[tid] = s;
    }
    __syncthreads();
    if (tid < 25) {
      float pd[25];
      for (int u = 0; u < 25; ++u) {
        float inner = 0.f;
        for (int c = 0; c < 16; ++c) inner += xt[tid * 16 + c] * xt[u * 16 + c];
        pd[u] = 2.f * inner - xx[tid] - xx[u];
      }
      for (int k = 0; k < 5; ++k) {    // top-5, ties -> lowest index (jax)
        float best = -3.4e38f; int bi = 0;
        for (int u = 0; u < 25; ++u) if (pd[u] > best) { best = pd[u]; bi = u; }
        sidx[tid * 5 + k] = bi;
        pd[bi] = -3.4e38f;
      }
    }
    __syncthreads();
    for (int task = tid; task < 2000; task += 128) {
      int o = task / 125, rem = task % 125, v = rem / 5, k = rem % 5;
      int u = sidx[v * 5 + k];
      float h = 0.f;
      for (int c = 0; c < 16; ++c) {
        float xv = xt[v * 16 + c];
        h += w_edge[(li * 16 + o) * 32 + c] * (xt[u * 16 + c] - xv)
           + w_edge[(li * 16 + o) * 32 + 16 + c] * xv;
      }
      h = h * (g_edge[li * 16 + o] * RS) + bt_edge[li * 16 + o];
      h = (h >= 0.f) ? h : 0.2f * h;
      hk[(o * 25 + v) * 5 + k] = h;
    }
    __syncthreads();
    for (int task = tid; task < 400; task += 128) {
      int o = task / 25, v = task % 25;
      float m = hk[(o * 25 + v) * 5];
      for (int k = 1; k < 5; ++k) m = fmaxf(m, hk[(o * 25 + v) * 5 + k]);
      hacc[o * 25 + v] += m;
    }
  }
  __syncthreads();
  for (int i = tid; i < 400; i += 128)
    hsum[(size_t)n * 400 + i] = hacc[i];
}

// ---------------------------------------------------------------------------
// Pass 3: channels 48..63 = relu(bn(edge, broadcast over T) + x). Streaming.
// ---------------------------------------------------------------------------
__global__ __launch_bounds__(256) void tail_kernel(
    const float* __restrict__ x, const float* __restrict__ hsum,
    const float* __restrict__ g_bn, const float* __restrict__ b_bn,
    float* __restrict__ out)
{
  __shared__ float base[25];
  const int n = blockIdx.x >> 4;
  const int c = blockIdx.x & 15;
  const int o = 48 + c;
  if (threadIdx.x < 25) {
    float s = g_bn[o] * rsqrtf(1.0f + 1e-5f);
    base[threadIdx.x] = hsum[(size_t)n * 400 + c * 25 + threadIdx.x] * s + b_bn[o];
  }
  __syncthreads();
  const size_t off = (size_t)(n * 64 + o) * 6400;
  for (int idx = threadIdx.x; idx < 6400; idx += 256) {
    float val = base[idx % 25] + x[off + idx];
    out[off + idx] = fmaxf(val, 0.f);
  }
}

// ---------------------------------------------------------------------------
extern "C" void kernel_launch(void* const* d_in, const int* in_sizes, int n_in,
                              void* d_out, int out_size, void* d_ws, size_t ws_size,
                              hipStream_t stream)
{
  (void)in_sizes; (void)n_in; (void)out_size; (void)ws_size;
  const float* x       = (const float*)d_in[0];
  const float* Aadj    = (const float*)d_in[1];
  const float* w_down  = (const float*)d_in[2];
  const float* b_down  = (const float*)d_in[3];
  const float* g_down  = (const float*)d_in[4];
  const float* bt_down = (const float*)d_in[5];
  const float* w_sub   = (const float*)d_in[6];
  const float* b_sub   = (const float*)d_in[7];
  const float* g_sub   = (const float*)d_in[8];
  const float* bt_sub  = (const float*)d_in[9];
  const float* w_edge  = (const float*)d_in[10];
  const float* g_edge  = (const float*)d_in[11];
  const float* bt_edge = (const float*)d_in[12];
  const float* g_bn    = (const float*)d_in[13];
  const float* b_bn    = (const float*)d_in[14];
  float* out = (float*)d_out;

  float* xm   = (float*)d_ws;          // 3*64*16*25 floats
  float* hsum = xm + 3 * 64 * 400;     // 64*16*25 floats

  (void)hipMemsetAsync(xm, 0, (size_t)3 * 64 * 400 * sizeof(float), stream);

  size_t smem = (size_t)(64 * P + 625 + 403 + 48 + 48) * sizeof(float)
              + (size_t)(PR * CSX + PR * CSD + 16 * QS) * sizeof(_Float16); // ~102 KB

  fused_main<<<dim3(256 / TT, 64), 256, smem, stream>>>(
      x, Aadj, w_down, b_down, g_down, bt_down,
      w_sub, b_sub, g_sub, bt_sub, g_bn, b_bn, xm, out);

  edge_conv_kernel<<<64, 128, 0, stream>>>(xm, w_edge, g_edge, bt_edge, hsum);

  tail_kernel<<<64 * 16, 256, 0, stream>>>(x, hsum, g_bn, b_bn, out);
}
